// EstimatorCV_18021682774195
// MI455X (gfx1250) — matrix operand order repaired and checked by reading, not compiled, not run on hardware
//
#include <hip/hip_runtime.h>

// ---------------------------------------------------------------------------
// EstimatorCV on MI455X (gfx1250):
//   segment_sum as one-hot GEMM on v_wmma_f32_16x16x32_f16 (f32 accumulate),
//   one-pass variance via  var = (sum(x^2) - ave*sum(x)) / amt,
//   K-split partials combined with f32 atomics, elementwise finalize.
// Round 2: amortize one-hot A-build over 2 N-tiles (32 cols/wave) and square
// in packed f16 to cut VALU-per-WMMA roughly in half.
// ---------------------------------------------------------------------------

typedef __attribute__((ext_vector_type(16))) _Float16 v16h;
typedef __attribute__((ext_vector_type(8)))  float    v8f;

#define NROWS   65536     // N
#define NF      512       // A (features)
#define NC      1000      // C (classes)
#define NCP     1024      // classes padded to multiple of 16
#define MBLK    64        // classes per wave = 4 WMMA M-tiles
#define NTILES  2         // 16-col N-tiles per wave (32 columns)
#define KCHUNKS 8
#define KCHUNK  (NROWS / KCHUNKS)   // 8192 rows per K-chunk

// Workspace layout (floats): sums[NCP*NF] | sumsq[NCP*NF] | n_c[NCP]

__global__ void zero_ws_kernel(float* __restrict__ ws, int n) {
    int i = blockIdx.x * blockDim.x + threadIdx.x;
    int stride = gridDim.x * blockDim.x;
    for (; i < n; i += stride) ws[i] = 0.0f;
}

__global__ void count_kernel(const int* __restrict__ labels, float* __restrict__ n_c) {
    int i = blockIdx.x * blockDim.x + threadIdx.x;
    if (i < NROWS) atomicAdd(&n_c[labels[i]], 1.0f);
}

// One wave per block. Each wave: 4 M-tiles (64 classes) x 2 N-tiles (32 cols)
// over one K-chunk of 8192 rows, stepping K by 32 per WMMA.
__global__ __launch_bounds__(32)
void wmma_accum_kernel(const float* __restrict__ X,
                       const int*   __restrict__ labels,
                       float* __restrict__ sums,
                       float* __restrict__ sumsq) {
    const int lane = threadIdx.x;       // 0..31
    const int half = lane >> 4;         // 0 = lanes 0-15, 1 = lanes 16-31
    const int lrow = lane & 15;

    int bid = blockIdx.x;
    const int mblk   = bid & 15;  bid >>= 4;   // 16 class blocks of 64
    const int npair  = bid & 15;  bid >>= 4;   // 16 column pairs of 32
    const int kchunk = bid;                    // 0..KCHUNKS-1

    const int classBase = mblk * MBLK;
    const int colBase   = npair * (16 * NTILES);
    const int kBase     = kchunk * KCHUNK;

    v8f accS[4][NTILES] = {};   // fp32 accumulators: sum(X) per (M-tile, N-tile)
    v8f accQ[4][NTILES] = {};   // sum(X^2)

    // B operand: lane holds column (colBase + nt*16 + lrow); K half by lane half.
    const float* bbase = X + (size_t)(kBase + half * 8) * NF + colBase + lrow;
    // A operand label stream: lanes<16 need K={0..7,16..23}, lanes>=16 K={8..15,24..31}.
    const int* lbase = labels + kBase + half * 8;

    for (int it = 0; it < KCHUNK / 32; ++it) {
        // ---- labels for this wave's 16 K-slots (A-matrix 16x32 f16 layout) ----
        const int* lp = lbase + it * 32;
        int lab[16];
#pragma unroll
        for (int j = 0; j < 8; ++j) {
            lab[j]     = lp[j];        // K = k0 + half*8 + j
            lab[8 + j] = lp[16 + j];   // K = k0 + 16 + half*8 + j
        }

        // ---- build one-hot A tiles for 4 consecutive 16-class groups ----
        v16h A0, A1, A2, A3;
#pragma unroll
        for (int e = 0; e < 16; ++e) {
            int rel = lab[e] - classBase;
            A0[e] = (rel == lrow     ) ? (_Float16)1.0f : (_Float16)0.0f;
            A1[e] = (rel == lrow + 16) ? (_Float16)1.0f : (_Float16)0.0f;
            A2[e] = (rel == lrow + 32) ? (_Float16)1.0f : (_Float16)0.0f;
            A3[e] = (rel == lrow + 48) ? (_Float16)1.0f : (_Float16)0.0f;
        }

        // ---- per N-tile: load B, square in packed f16, 8 WMMAs ----
        const float* bp0 = bbase + (size_t)it * 32 * NF;
#pragma unroll
        for (int nt = 0; nt < NTILES; ++nt) {
            v16h B, B2;
            const float* bp = bp0 + nt * 16;
#pragma unroll
            for (int e = 0; e < 16; ++e) {
                B[e] = (_Float16)bp[(size_t)(e + (e & 8)) * NF];
            }
#pragma unroll
            for (int e = 0; e < 16; ++e) {
                B2[e] = B[e] * B[e];          // v_pk_mul_f16
            }
            accS[0][nt] = __builtin_amdgcn_wmma_f32_16x16x32_f16(false, A0, false, B,  (short)0, accS[0][nt], false, false);
            accQ[0][nt] = __builtin_amdgcn_wmma_f32_16x16x32_f16(false, A0, false, B2, (short)0, accQ[0][nt], false, false);
            accS[1][nt] = __builtin_amdgcn_wmma_f32_16x16x32_f16(false, A1, false, B,  (short)0, accS[1][nt], false, false);
            accQ[1][nt] = __builtin_amdgcn_wmma_f32_16x16x32_f16(false, A1, false, B2, (short)0, accQ[1][nt], false, false);
            accS[2][nt] = __builtin_amdgcn_wmma_f32_16x16x32_f16(false, A2, false, B,  (short)0, accS[2][nt], false, false);
            accQ[2][nt] = __builtin_amdgcn_wmma_f32_16x16x32_f16(false, A2, false, B2, (short)0, accQ[2][nt], false, false);
            accS[3][nt] = __builtin_amdgcn_wmma_f32_16x16x32_f16(false, A3, false, B,  (short)0, accS[3][nt], false, false);
            accQ[3][nt] = __builtin_amdgcn_wmma_f32_16x16x32_f16(false, A3, false, B2, (short)0, accQ[3][nt], false, false);
        }
    }

    // ---- combine K-chunk partials (C/D layout: M = reg + 8*half, N = lrow) ----
#pragma unroll
    for (int t = 0; t < 4; ++t) {
#pragma unroll
        for (int nt = 0; nt < NTILES; ++nt) {
#pragma unroll
            for (int r = 0; r < 8; ++r) {
                int row = classBase + t * 16 + r + half * 8;
                int col = colBase + nt * 16 + lrow;
                atomicAdd(&sums [row * NF + col], accS[t][nt][r]);
                atomicAdd(&sumsq[row * NF + col], accQ[t][nt][r]);
            }
        }
    }
}

__global__ void finalize_kernel(const float* __restrict__ sums,
                                const float* __restrict__ sumsq,
                                const float* __restrict__ n_c,
                                const float* __restrict__ count,
                                const float* __restrict__ mean,
                                const float* __restrict__ cov,
                                float* __restrict__ out) {
    int idx = blockIdx.x * blockDim.x + threadIdx.x;
    if (idx >= NC * NF) return;
    int c = idx / NF;
    int a = idx - c * NF;

    float nc  = n_c[c];
    float cnt = count[c];
    float s   = sums [c * NF + a];
    float q   = sumsq[c * NF + a];

    float amt = (nc == 0.0f) ? 1.0f : nc;       // Amount (zeros -> 1)
    float ave = s / amt;
    float var = (q - ave * s) / amt;            // == segsum((x-ave)^2)/amt

    float denom = nc + cnt;
    float w = (denom == 0.0f) ? 0.0f : (nc / denom);
    float omw = 1.0f - w;

    float m  = mean[idx];
    float cv = cov[idx];
    float d  = m - ave;

    out[idx]                = cv * omw + var * w + w * omw * d * d;  // cov_new
    out[NC * NF + idx]      = m * omw + ave * w;                     // mean_new
    if (a == 0) out[2 * NC * NF + c] = cnt + nc;                     // count_new
}

extern "C" void kernel_launch(void* const* d_in, const int* in_sizes, int n_in,
                              void* d_out, int out_size, void* d_ws, size_t ws_size,
                              hipStream_t stream) {
    const float* X      = (const float*)d_in[0];   // features [N, A] f32
    const int*   labels = (const int*)  d_in[1];   // labels   [N]    i32
    const float* count  = (const float*)d_in[2];   // count    [C]    f32
    const float* mean   = (const float*)d_in[3];   // mean     [C, A] f32
    const float* cov    = (const float*)d_in[4];   // cov      [C, A] f32
    float* out = (float*)d_out;                    // cov_new | mean_new | count_new
    float* ws  = (float*)d_ws;

    float* sums  = ws;
    float* sumsq = ws + (size_t)NCP * NF;
    float* nvec  = ws + (size_t)2 * NCP * NF;

    const int zn = 2 * NCP * NF + NCP;
    zero_ws_kernel<<<512, 256, 0, stream>>>(ws, zn);
    count_kernel<<<NROWS / 256, 256, 0, stream>>>(labels, nvec);
    wmma_accum_kernel<<<16 * 16 * KCHUNKS, 32, 0, stream>>>(X, labels, sums, sumsq);
    finalize_kernel<<<(NC * NF + 255) / 256, 256, 0, stream>>>(sums, sumsq, nvec,
                                                               count, mean, cov, out);
}
